// EmbeddingNet_3d_new_big_49271864819835
// MI455X (gfx1250) — matrix operand order repaired and checked by reading, not compiled, not run on hardware
//
#include <hip/hip_runtime.h>
#include <math.h>

typedef __attribute__((ext_vector_type(16))) __bf16 v16bf;
typedef __attribute__((ext_vector_type(8)))  float  v8f;

#define TPB 256
#define ROWS_PER_WG 64

// ---- dynamic LDS layout -------------------------------------------------
// eT  : f32  [64][416]          e tile (stats); only cols 0..409 used
// a1s : bf16 [4][13][32][16]    A fragments GEMM1 (swizzled, K pad zeroed)
// w2sw: bf16 [2][14][32][16]    B fragments GEMM1, double-buffered K-slab
// a2s : bf16 [4][7][32][16]     A fragments GEMM2 (= relu(e@W2+b2))
// w3sw: bf16 [4][7][32][16]     B fragments GEMM2 (W3)
// h2  : f32  [64][64]           cols 0..49 relu(h1@W3+b3), 50..63 e_arti
// art : f32  [64]
constexpr int ET_S = 416;

constexpr size_t OFF_ET  = 0;                                      // 106496
constexpr size_t OFF_A1S = OFF_ET  + (size_t)64 * ET_S * 4;        // +53248
constexpr size_t OFF_W2S = OFF_A1S + (size_t)4 * 13 * 32 * 16 * 2; // +28672
constexpr size_t OFF_A2S = OFF_W2S + (size_t)2 * 14 * 32 * 16 * 2; // +28672
constexpr size_t OFF_W3S = OFF_A2S + (size_t)4 * 7 * 32 * 16 * 2;  // +28672
constexpr size_t OFF_H2  = OFF_W3S + (size_t)4 * 7 * 32 * 16 * 2;  // +16384
constexpr size_t OFF_ART = OFF_H2  + (size_t)64 * 64 * 4;          // +256
constexpr size_t SMEM_BYTES = OFF_ART + 64 * 4;                    // 262400

constexpr int W2_BUF_ELEMS = 14 * 32 * 16;   // 7168 bf16 per slab buffer

__device__ __forceinline__ float rsum4(float v) {
    v += __shfl_xor(v, 1); v += __shfl_xor(v, 2); return v;
}
__device__ __forceinline__ float rmax4(float v) {
    v = fmaxf(v, __shfl_xor(v, 1)); v = fmaxf(v, __shfl_xor(v, 2)); return v;
}
__device__ __forceinline__ unsigned short bf16bits(float f) {
    return __builtin_bit_cast(unsigned short, (__bf16)f);
}

__global__ __launch_bounds__(TPB)
void embnet_fused_wmma(const float* __restrict__ e,
                       const float* __restrict__ W2, const float* __restrict__ b2,
                       const float* __restrict__ W3, const float* __restrict__ b3,
                       const float* __restrict__ W4, const float* __restrict__ b4,
                       float* __restrict__ out, int nrows)
{
    extern __shared__ unsigned char smem[];
    float*  eT   = (float*) (smem + OFF_ET);
    __bf16* a1s  = (__bf16*)(smem + OFF_A1S);
    __bf16* w2sw = (__bf16*)(smem + OFF_W2S);
    __bf16* a2s  = (__bf16*)(smem + OFF_A2S);
    __bf16* w3sw = (__bf16*)(smem + OFF_W3S);
    float*  h2   = (float*) (smem + OFF_H2);
    float*  artA = (float*) (smem + OFF_ART);

    const int tid  = threadIdx.x;
    const int lane = tid & 31;
    const int wave = tid >> 5;
    const int half = lane >> 4;
    const int ln   = lane & 15;
    const int row0 = blockIdx.x * ROWS_PER_WG;

    // ---- Phase 0: zero K-padded fragment arrays (b128 stores) ----------
    {
        uint4 z{0u, 0u, 0u, 0u};
        uint4* za = (uint4*)a1s;                     // 53248/16 = 3328
        for (int i = tid; i < 3328; i += TPB) za[i] = z;
        uint4* zb = (uint4*)a2s;                     // 28672/16 = 1792
        for (int i = tid; i < 1792; i += TPB) zb[i] = z;
        uint4* zc = (uint4*)w3sw;                    // 1792
        for (int i = tid; i < 1792; i += TPB) zc[i] = z;
    }
    __syncthreads();

    // ---- Phase 1: stage eT + A1 (float2, packed b32 scatter), W3 -------
    for (int i = tid; i < (64 * 410) / 2; i += TPB) {
        int flat = i * 2;
        int r = flat / 410, c = flat - r * 410;      // c even
        int gr = row0 + r;
        float2 v = make_float2(0.f, 0.f);
        if (gr < nrows) v = *(const float2*)(e + (size_t)gr * 410 + c);
        eT[r * ET_S + c]     = v.x;
        eT[r * ET_S + c + 1] = v.y;
        if (gr < nrows) {
            int mt = r >> 4, rit = r & 15;
            int ks = c >> 5, kk = c & 31;            // kk even
            int l2 = rit + ((kk >> 4) << 4);
            int j  = kk & 15;                        // even; pair is (j, j+1)
            unsigned int pk = (unsigned int)bf16bits(v.x) |
                              ((unsigned int)bf16bits(v.y) << 16);
            *(unsigned int*)(a1s + ((mt * 13 + ks) * 32 + l2) * 16 + j) = pk;
        }
    }
    for (int i = tid; i < (200 * 50) / 2; i += TPB) {
        int flat = i * 2;
        int k = flat / 50, n = flat - k * 50;        // n even, pair same row
        float2 v = *(const float2*)(W3 + flat);
        int nt = n >> 4, ks = k >> 5, kk = k & 31;
        int base = ((nt * 7 + ks) * 32 + (n & 15) + ((kk >> 4) << 4)) * 16 + (kk & 15);
        w3sw[base]      = (__bf16)v.x;
        w3sw[base + 16] = (__bf16)v.y;
    }
    __syncthreads();

    // ---- Phase 2: per-row statistics (4 lanes per row, from eT) --------
    {
        const int r    = tid >> 2;
        const int part = tid & 3;
        const float* E = eT + r * ET_S;

        float s1 = 0.f, q1 = 0.f, s2 = 0.f, q2 = 0.f, d2 = 0.f;
        for (int c = part; c < 108; c += 4) {
            float x = E[98 + c], y = E[302 + c];
            s1 += x; q1 += x * x; s2 += y; q2 += y * y;
            float d = x - y; d2 += d * d;
        }
        s1 = rsum4(s1); q1 = rsum4(q1); s2 = rsum4(s2); q2 = rsum4(q2); d2 = rsum4(d2);
        float mp1 = s1 * (1.f / 108.f), mp2 = s2 * (1.f / 108.f);
        float vp1 = (q1 - s1 * s1 * (1.f / 108.f)) * (1.f / 107.f);
        float vp2 = (q2 - s2 * s2 * (1.f / 108.f)) * (1.f / 107.f);
        float msd = d2 * (1.f / 108.f);

        float mad[3], mx1[3], mx2[3], v1[3], v2[3];
        #pragma unroll
        for (int ch = 0; ch < 3; ++ch) {
            const int lo1 = 2 + 32 * ch, lo2 = 206 + 32 * ch;
            float sa = 0.f, sx = 0.f, qx = 0.f, sy = 0.f, qy = 0.f;
            float m1 = -3.4e38f, m2 = -3.4e38f;
            for (int c = part; c < 32; c += 4) {
                float x = E[lo1 + c], y = E[lo2 + c];
                sa += fabsf(x - y);
                sx += x; qx += x * x; sy += y; qy += y * y;
                m1 = fmaxf(m1, x); m2 = fmaxf(m2, y);
            }
            sa = rsum4(sa); sx = rsum4(sx); qx = rsum4(qx);
            sy = rsum4(sy); qy = rsum4(qy);
            m1 = rmax4(m1); m2 = rmax4(m2);
            mad[ch] = sa * (1.f / 32.f);
            mx1[ch] = m1; mx2[ch] = m2;
            v1[ch] = (qx - sx * sx * (1.f / 32.f)) * (1.f / 31.f);
            v2[ch] = (qy - sy * sy * (1.f / 32.f)) * (1.f / 31.f);
        }

        float sA = 0.f, qA = 0.f, sB = 0.f, qB = 0.f;
        for (int c = 2 + part; c < 194; c += 4)   { float x = E[c]; sA += x; qA += x * x; }
        for (int c = 194 + part; c < 410; c += 4) { float x = E[c]; sB += x; qB += x * x; }
        sA = rsum4(sA); qA = rsum4(qA); sB = rsum4(sB); qB = rsum4(qB);
        float varA = (qA - sA * sA * (1.f / 192.f)) * (1.f / 191.f);
        float varB = (qB - sB * sB * (1.f / 216.f)) * (1.f / 215.f);
        float ang = E[0];

        if (part == 0) {
            float* H = h2 + r * 64 + 50;
            H[0]  = 0.1f * mp1;  H[1]  = 0.1f * mp2;
            H[2]  = 5.f * vp1;   H[3]  = 5.f * vp2;
            H[4]  = msd;
            H[5]  = 20.f * mad[0]; H[6] = 20.f * mad[1]; H[7] = 20.f * mad[2];
            H[8]  = 5.f * mx1[0]; H[9]  = 5.f * mx2[0];
            H[10] = 5.f * mx1[1]; H[11] = 5.f * mx2[1];
            H[12] = 5.f * mx1[2]; H[13] = 5.f * mx2[2];

            const float PIc = 3.14159f;
            float t = 0.f;
            t += ((mp1 < 3.f/256.f) || (mp2 < 3.f/256.f) ||
                  (mp1 > 253.f/256.f) || (mp2 > 253.f/256.f)) ? 30.f : 0.f;
            t += ((varA < 0.01f) || (varB < 0.01f)) ? 1.f : 0.f;
            t += ang;
            t += (ang < 20.f * PIc / 180.f) ? 0.1f : 0.f;
            t += ((ang < 30.f * PIc / 180.f) && (msd < 0.001f)) ? 0.1f : 0.f;
            t += ((ang < 90.f * PIc / 180.f) && (msd < 0.001f)) ? 10.f : 0.f;
            t += ((msd > 0.15f) || (msd < 0.0003f)) ? 0.3f : 0.f;
            t += 50.f * (mad[0] + mad[1] + mad[2]);
            t += ((mad[0] > 0.003f) || (mad[1] > 0.003f) || (mad[2] > 0.003f)) ? 0.1f : 0.f;
            bool mx03 = (mx1[0] > 0.03f) || (mx2[0] > 0.03f) || (mx1[1] > 0.03f) ||
                        (mx2[1] > 0.03f) || (mx1[2] > 0.03f) || (mx2[2] > 0.03f);
            bool mx06 = (mx1[0] > 0.06f) || (mx2[0] > 0.06f) || (mx1[1] > 0.06f) ||
                        (mx2[1] > 0.06f) || (mx1[2] > 0.06f) || (mx2[2] > 0.06f);
            t += mx03 ? 0.1f : 0.f;
            t += mx06 ? 3.f : 0.f;
            t += 30.f * (v1[0] + v1[1] + v1[2] + v2[0] + v2[1] + v2[2]);
            artA[r] = t;
        }
    }

    // ---- Phase 3: GEMM1, uniform 7 n-tiles/wave, double-buffered -------
    {
        const int mtile = __builtin_amdgcn_readfirstlane(wave & 3);
        const int nt0   = __builtin_amdgcn_readfirstlane((wave >> 2) * 7);
        v8f acc[7];
        #pragma unroll
        for (int i = 0; i < 7; ++i) acc[i] = v8f{0.f,0.f,0.f,0.f,0.f,0.f,0.f,0.f};

        // W2 slab = contiguous 6400 floats; stage via float4 (no row cross)
        auto stageW2 = [&](int ks, int buf) {
            __bf16* dst = w2sw + buf * W2_BUF_ELEMS;
            for (int i = tid; i < 1600; i += TPB) {
                if (ks * 1600 + i < 20500) {          // 82000/4 total float4s
                    float4 v = *(const float4*)(W2 + (size_t)ks * 6400 + (size_t)i * 4);
                    int fl = i * 4;
                    int kk = fl / 200;                // 0..31 (row within slab)
                    int n0 = fl - kk * 200;           // multiple of 4
                    int l2b = (kk >> 4) << 4;
                    int jj  = kk & 15;
                    float vv[4] = {v.x, v.y, v.z, v.w};
                    #pragma unroll
                    for (int q = 0; q < 4; ++q) {
                        int n = n0 + q;
                        dst[(((n >> 4) * 32 + (n & 15) + l2b) << 4) + jj] = (__bf16)vv[q];
                    }
                }
            }
        };

        stageW2(0, 0);
        __syncthreads();

        for (int ks = 0; ks < 13; ++ks) {
            const int buf = ks & 1;
            if (ks + 1 < 13) {
                if (ks + 2 < 13) {                    // 1 cacheline/thread ahead
                    int off = (ks + 2) * 6400 + tid * 32;
                    if (off < 82000) __builtin_prefetch(W2 + off, 0, 1);
                }
                stageW2(ks + 1, buf ^ 1);
            }
            const __bf16* w2b = w2sw + buf * W2_BUF_ELEMS;
            v16bf a = *(const v16bf*)(a1s + ((mtile * 13 + ks) * 32 + lane) * 16);
            v16bf bf[7];
            #pragma unroll
            for (int t = 0; t < 7; ++t)
                bf[t] = *(const v16bf*)(w2b + ((nt0 + t) * 32 + lane) * 16);
            #pragma unroll
            for (int t = 0; t < 7; ++t)
                acc[t] = __builtin_amdgcn_wmma_f32_16x16x32_bf16(
                    false, a, false, bf[t], (short)0, acc[t], false, false);
            __syncthreads();
        }

        // D -> a2s (bias+relu) scattered into GEMM2 A-fragment layout
        #pragma unroll
        for (int t = 0; t < 7; ++t) {
            int n = (nt0 + t) * 16 + ln;
            if (n < 200) {
                float bias = b2[n];
                int ks2 = n >> 5, kk = n & 31;
                int j2  = kk & 15;
                int lhi = (kk >> 4) << 4;
                #pragma unroll
                for (int rr = 0; rr < 8; ++rr) {
                    int rit = half * 8 + rr;
                    float v = fmaxf(acc[t][rr] + bias, 0.f);
                    a2s[((mtile * 7 + ks2) * 32 + rit + lhi) * 16 + j2] = (__bf16)v;
                }
            }
        }
    }
    __syncthreads();

    // ---- Phase 4: GEMM2  h2[:, :50] = relu(h1 @ W3 + b3) ---------------
    {
        const int mt  = __builtin_amdgcn_readfirstlane(wave & 3);
        const int ntA = __builtin_amdgcn_readfirstlane(wave >> 2);  // tiles {ntA, ntA+2}
        v8f acc2[2];
        acc2[0] = v8f{0.f,0.f,0.f,0.f,0.f,0.f,0.f,0.f};
        acc2[1] = acc2[0];

        for (int ks = 0; ks < 7; ++ks) {
            v16bf a  = *(const v16bf*)(a2s  + ((mt * 7 + ks) * 32 + lane) * 16);
            v16bf b0 = *(const v16bf*)(w3sw + ((ntA * 7 + ks) * 32 + lane) * 16);
            v16bf b1 = *(const v16bf*)(w3sw + (((ntA + 2) * 7 + ks) * 32 + lane) * 16);
            acc2[0] = __builtin_amdgcn_wmma_f32_16x16x32_bf16(
                false, a, false, b0, (short)0, acc2[0], false, false);
            acc2[1] = __builtin_amdgcn_wmma_f32_16x16x32_bf16(
                false, a, false, b1, (short)0, acc2[1], false, false);
        }
        #pragma unroll
        for (int u = 0; u < 2; ++u) {
            int n = (ntA + 2 * u) * 16 + ln;
            if (n < 50) {
                float bias = b3[n];
                #pragma unroll
                for (int rr = 0; rr < 8; ++rr) {
                    int m = mt * 16 + half * 8 + rr;
                    h2[m * 64 + n] = fmaxf(acc2[u][rr] + bias, 0.f);
                }
            }
        }
    }
    __syncthreads();

    // ---- Phase 5: w_emb = h2 @ W4 + b4 ; out = exp(-.5*art + .5*w_emb) -
    if (tid < 64) {
        int gr = row0 + tid;
        if (gr < nrows) {
            float s = 0.f;
            const float* H = h2 + tid * 64;
            #pragma unroll 8
            for (int c = 0; c < 64; ++c) s += H[c] * W4[c];
            float w_emb = s + b4[0];
            out[gr] = expf(-0.5f * artA[tid] + 0.5f * w_emb);
        }
    }
}

extern "C" void kernel_launch(void* const* d_in, const int* in_sizes, int n_in,
                              void* d_out, int out_size, void* d_ws, size_t ws_size,
                              hipStream_t stream) {
    const float* e  = (const float*)d_in[0];
    const float* W2 = (const float*)d_in[1];
    const float* b2 = (const float*)d_in[2];
    const float* W3 = (const float*)d_in[3];
    const float* b3 = (const float*)d_in[4];
    const float* W4 = (const float*)d_in[5];
    const float* b4 = (const float*)d_in[6];
    float* out = (float*)d_out;
    (void)d_ws; (void)ws_size; (void)n_in; (void)out_size;

    int rows = in_sizes[0] / 410;
    int grid = (rows + ROWS_PER_WG - 1) / ROWS_PER_WG;

    hipFuncSetAttribute(reinterpret_cast<const void*>(embnet_fused_wmma),
                        hipFuncAttributeMaxDynamicSharedMemorySize,
                        (int)SMEM_BYTES);
    embnet_fused_wmma<<<grid, TPB, SMEM_BYTES, stream>>>(
        e, W2, b2, W3, b3, W4, b4, out, rows);
}
